// CvT_Block_63548336111634
// MI455X (gfx1250) — matrix operand order repaired and checked by reading, not compiled, not run on hardware
//
#include <hip/hip_runtime.h>
#include <math.h>

typedef _Float16 f16;
typedef __attribute__((ext_vector_type(16))) _Float16 v16h;
typedef __attribute__((ext_vector_type(8)))  float    v8f;

#define NPOS 1024
#define CDIM 128
#define HEADS 8
#define DKDIM 32
#define INV_SQRT2F 0.70710678118654752f
#define INV_SQRT_DK 0.17677669529663687f  // 1/sqrt(32)

// ---------------- WMMA helpers (16x16x32 f16 -> f32) ----------------
// A-matrix 16-bit layout: lane = m + 16*s (m=row, s=half-select);
//   elems 0..7  -> K = 8*s + e        (contiguous 8 halves)
//   elems 8..15 -> K = 16 + 8*s + e   (contiguous 8 halves)
__device__ inline v16h load_a(const f16* A, int row0, int ld, int k0, int lane) {
  int m = lane & 15, s = lane >> 4;
  const f16* p = A + (size_t)(row0 + m) * ld + k0 + 8 * s;
  v16h a;
#pragma unroll
  for (int e = 0; e < 8; ++e) a[e] = p[e];
#pragma unroll
  for (int e = 0; e < 8; ++e) a[8 + e] = p[16 + e];
  return a;
}

// B-matrix 16-bit layout: lane = n + 16*s; elems 0..7 -> K = 8*s + e, elems 8..15 -> K = 16 + 8*s + e.
// Source "Bt" stored N-major with contiguous K (Bt[n][k], row stride ld).
__device__ inline v16h load_b(const f16* Bt, int n0, int ld, int k0, int lane) {
  int n = lane & 15, s = lane >> 4;
  const f16* p = Bt + (size_t)(n0 + n) * ld + k0 + 8 * s;
  v16h b;
#pragma unroll
  for (int e = 0; e < 8; ++e) b[e] = p[e];
#pragma unroll
  for (int e = 0; e < 8; ++e) b[8 + e] = p[16 + e];
  return b;
}

__device__ inline v8f wmma_f16(v16h a, v16h b, v8f c) {
  return __builtin_amdgcn_wmma_f32_16x16x32_f16(false, a, false, b, (short)0, c,
                                                false, false);
}

// C/D layout: VGPR g holds row M = g + 8*(lane>>4), col N = lane&15.

__device__ inline int relidx(int i, int j) {
  int dx = ((i >> 5) - (j >> 5)) & 31;
  int dy = ((i & 31) - (j & 31)) & 31;
  return (dx << 5) | dy;
}

// Async-stage one contiguous 1KB block (16 rows x 64B) from global to LDS.
// Each of 32 lanes moves 16B per issue; INST_OFFSET applies to both the
// global and LDS address, so offset:512 covers the second 512B chunk.
// Increments ASYNCcnt by 2 (pairs complete in issue order).
__device__ inline void async_stage_1kb(const void* gsrc, unsigned ldsOff,
                                       int lane) {
  unsigned long long ga = (unsigned long long)(size_t)gsrc + (unsigned)(lane * 16);
  unsigned la = ldsOff + (unsigned)(lane * 16);
  asm volatile("global_load_async_to_lds_b128 %0, %1, off"
               :: "v"(la), "v"(ga) : "memory");
  asm volatile("global_load_async_to_lds_b128 %0, %1, off offset:512"
               :: "v"(la), "v"(ga) : "memory");
}

// ---------------- elementwise / conv kernels (VALU, negligible FLOPs) -----

__global__ __launch_bounds__(256) void cvt_kernel(
    const float* RQ, const float* RK, const float* RV, const float* Wq,
    const float* W0, f16* RQt, f16* RKt, f16* RVh, f16* Wqh, f16* W0h) {
  int idx = blockIdx.x * 256 + threadIdx.x;  // 32768 total
  int d = idx >> 10, r = idx & 1023;
  RQt[r * DKDIM + d] = (f16)RQ[idx];  // [r][d] transposed
  RKt[r * DKDIM + d] = (f16)RK[idx];
  RVh[idx] = (f16)RV[idx];            // [d][r] natural
  Wqh[idx] = (f16)Wq[idx];            // 256*128 == 32768
  W0h[idx] = (f16)W0[idx];            // 128*256 == 32768
}

__global__ __launch_bounds__(256) void gelu_kernel(const float* x, float* gx) {
  int idx = blockIdx.x * 256 + threadIdx.x;
  float u = x[idx] * INV_SQRT2F;
  float t = tanhf(0.7978845608028654f * (u + 0.044715f * u * u * u));
  gx[idx] = 0.5f * u * (1.0f + t);
}

__global__ __launch_bounds__(256) void embed_kernel(
    const float* gx, const float* W, const float* b, const float* x, float* h,
    f16* h2t) {
  int idx = blockIdx.x * 256 + threadIdx.x;  // 131072
  int co = idx >> 10, i = idx & 1023;
  int xi = i >> 5, yi = i & 31;
  float acc = b[co];
  for (int ci = 0; ci < CDIM; ++ci) {
    const float* wp = W + (size_t)(co * CDIM + ci) * 9;
    const float* gp = gx + ci * NPOS;
#pragma unroll
    for (int ky = 0; ky < 3; ++ky) {
      int sx = (xi + ky - 1) & 31;
#pragma unroll
      for (int kx = 0; kx < 3; ++kx) {
        int sy = (yi + kx - 1) & 31;
        acc += gp[(sx << 5) | sy] * wp[ky * 3 + kx];
      }
    }
  }
  float hv = acc + x[idx];
  h[idx] = hv;                                  // residual (f32, [c][i])
  h2t[i * CDIM + co] = (f16)(hv * INV_SQRT2F);  // scaled, [i][c]
}

__global__ __launch_bounds__(256) void dw_kernel(const float* h,
                                                 const float* Wkd,
                                                 const float* Wvd, float* hk,
                                                 float* hv) {
  int idx = blockIdx.x * 256 + threadIdx.x;  // 131072
  int c = idx >> 10, i = idx & 1023;
  int xi = i >> 5, yi = i & 31;
  float ak = 0.f, av = 0.f;
#pragma unroll
  for (int ky = 0; ky < 3; ++ky) {
    int sx = (xi + ky - 1) & 31;
#pragma unroll
    for (int kx = 0; kx < 3; ++kx) {
      int sy = (yi + kx - 1) & 31;
      float v = h[c * NPOS + ((sx << 5) | sy)] * INV_SQRT2F;
      ak += v * Wkd[c * 9 + ky * 3 + kx];
      av += v * Wvd[c * 9 + ky * 3 + kx];
    }
  }
  hk[idx] = ak;
  hv[idx] = av;
}

__global__ __launch_bounds__(256) void pw_kernel(const float* hk,
                                                 const float* hv,
                                                 const float* Wkp,
                                                 const float* Wvp, f16* Kt,
                                                 f16* Vh) {
  int idx = blockIdx.x * 256 + threadIdx.x;  // 262144
  int ch = idx >> 10, i = idx & 1023;
  int hh = ch >> 5, d = ch & 31;
  float ak = 0.f, av = 0.f;
#pragma unroll
  for (int c = 0; c < 16; ++c) {
    int src = (hh * 16 + c) * NPOS + i;
    ak += Wkp[ch * 16 + c] * hk[src];
    av += Wvp[ch * 16 + c] * hv[src];
  }
  Kt[(size_t)hh * NPOS * DKDIM + i * DKDIM + d] = (f16)ak;  // [h][i][d]
  Vh[(size_t)hh * DKDIM * NPOS + d * NPOS + i] = (f16)av;   // [h][d][i]
}

// ---------------- WMMA GEMM kernels ----------------

// Q = Wq(256x128) @ h2(128x1024), scattered into Qt[h][i][d]
__global__ __launch_bounds__(256) void qproj_kernel(const f16* Wqh,
                                                    const f16* h2t, f16* Qt) {
  int tileId = blockIdx.x * 8 + (threadIdx.x >> 5);  // 1024 tiles (16 x 64)
  int lane = threadIdx.x & 31;
  int m0 = (tileId >> 6) * 16, i0 = (tileId & 63) * 16;
  v8f acc = {};
#pragma unroll
  for (int kc = 0; kc < 4; ++kc) {
    v16h a = load_a(Wqh, m0, CDIM, kc * 32, lane);
    v16h b = load_b(h2t, i0, CDIM, kc * 32, lane);
    acc = wmma_f16(a, b, acc);
  }
  int n = lane & 15, s = lane >> 4;
  int i = i0 + n;
#pragma unroll
  for (int g = 0; g < 8; ++g) {
    int ch = m0 + g + 8 * s;
    Qt[(size_t)(ch >> 5) * NPOS * DKDIM + i * DKDIM + (ch & 31)] = (f16)acc[g];
  }
}

// T[h][m][r] = Xt[h](1024x32) @ Rt(1024x32)^T  (single K=32 chunk per tile)
__global__ __launch_bounds__(256) void tgemm_kernel(const f16* Xt,
                                                    const f16* Rt, f16* T) {
  int tileId = blockIdx.x * 8 + (threadIdx.x >> 5);  // 32768 tiles
  int lane = threadIdx.x & 31;
  int h = tileId >> 12;
  int t = tileId & 4095;
  int m0 = (t >> 6) * 16, r0 = (t & 63) * 16;
  v16h a = load_a(Xt + (size_t)h * NPOS * DKDIM, m0, DKDIM, 0, lane);
  v16h b = load_b(Rt, r0, DKDIM, 0, lane);
  v8f c = {};
  c = wmma_f16(a, b, c);
  int n = lane & 15, s = lane >> 4;
  f16* Th = T + (size_t)h * NPOS * NPOS;
#pragma unroll
  for (int g = 0; g < 8; ++g)
    Th[(size_t)(m0 + g + 8 * s) * NPOS + r0 + n] = (f16)c[g];
}

// Fused attention per (head, 16-row i-tile): async-staged K tiles, score WMMA
// + iRPE gathers + softmax + alpha@V^T + alphaPerm@RV^T (4-wave split + LDS
// reduction), write att_t[i][h*32+d] (f16).
__global__ __launch_bounds__(256) void attn_kernel(
    const f16* Qt, const f16* Kt, const f16* Vh, const f16* TK, const f16* TQ,
    const f16* RVh, f16* att_t) {
  __shared__ float S[16][NPOS];          // 64 KB  scores (f32 for softmax)
  __shared__ f16 A16[16][NPOS];          // 32 KB  alpha
  __shared__ f16 AP16[16][NPOS];         // 32 KB  alpha permuted by rel(i,j)
  __shared__ float red[16][16];          // 1 KB   row reductions
  __shared__ f16 KBUF[8][2][16][DKDIM];  // 16 KB  double-buffered K tiles
  __shared__ float Opart[2][16][16];     // 2 KB   phase-3 partial tiles

  int h = blockIdx.x >> 6;
  int i0 = (blockIdx.x & 63) * 16;
  int tid = threadIdx.x;
  int wave = tid >> 5, lane = tid & 31;
  int n = lane & 15, s = lane >> 4;

  const f16* Qh = Qt + (size_t)h * NPOS * DKDIM;
  const f16* Kh = Kt + (size_t)h * NPOS * DKDIM;
  const f16* TKh = TK + (size_t)h * NPOS * NPOS;
  const f16* TQh = TQ + (size_t)h * NPOS * NPOS;

  // Phase 1: S[i_local][j] = (Q_i.K_j + TK[i,rel] + TQ[j,rel]) / sqrt(dk)
  // K j-tiles are contiguous 1KB blocks in Kt[h][j][d]; stage them into LDS
  // with async-to-LDS copies, double-buffered per wave on ASYNCcnt.
  {
    v16h a = load_a(Qh, i0, DKDIM, 0, lane);  // Q tile reused across j-tiles
    unsigned buf0 = (unsigned)(size_t)&KBUF[wave][0][0][0];
    unsigned buf1 = (unsigned)(size_t)&KBUF[wave][1][0][0];
    async_stage_1kb(Kh + (size_t)wave * 16 * DKDIM, buf0, lane);
    for (int idx = 0; idx < 8; ++idx) {
      int jt = wave + idx * 8;
      int cur = idx & 1;
      if (idx < 7) {
        async_stage_1kb(Kh + (size_t)(jt + 8) * 16 * DKDIM, cur ? buf0 : buf1,
                        lane);
        asm volatile("s_wait_asynccnt 2" ::: "memory");  // tile idx ready
      } else {
        asm volatile("s_wait_asynccnt 0" ::: "memory");
      }
      v16h b = load_b((const f16*)&KBUF[wave][cur][0][0], 0, DKDIM, 0, lane);
      v8f c = {};
      c = wmma_f16(a, b, c);
      int j0 = jt * 16;
      int j = j0 + n;
#pragma unroll
      for (int g = 0; g < 8; ++g) {
        int mi = g + 8 * s;
        int i = i0 + mi;
        int r = relidx(i, j);
        float v = c[g] + (float)TKh[(size_t)i * NPOS + r] +
                  (float)TQh[(size_t)j * NPOS + r];
        S[mi][j] = v * INV_SQRT_DK;
      }
    }
  }
  __syncthreads();

  // Phase 2: row softmax (16 threads per row), write A16 and AP16
  {
    int row = tid >> 4, col = tid & 15;
    float m = -1e30f;
    for (int j = col; j < NPOS; j += 16) m = fmaxf(m, S[row][j]);
    red[row][col] = m;
    __syncthreads();
    float rm = red[row][0];
#pragma unroll
    for (int k = 1; k < 16; ++k) rm = fmaxf(rm, red[row][k]);
    __syncthreads();
    float sum = 0.f;
    for (int j = col; j < NPOS; j += 16) {
      float e = __expf(S[row][j] - rm);
      S[row][j] = e;
      sum += e;
    }
    red[row][col] = sum;
    __syncthreads();
    float rs = 0.f;
#pragma unroll
    for (int k = 0; k < 16; ++k) rs += red[row][k];
    float inv = 1.0f / rs;
    int ig = i0 + row;
    for (int j = col; j < NPOS; j += 16) {
      float av = S[row][j] * inv;
      f16 ah = (f16)av;
      A16[row][j] = ah;
      AP16[row][relidx(ig, j)] = ah;  // per-row bijection: no conflicts
    }
  }
  __syncthreads();

  // Phase 3: O = A16 @ V^T + AP16 @ RV^T.
  // Waves 0,1: alpha@V (d-tiles 0,1). Waves 2,3: alphaPerm@RV (d-tiles 0,1)
  // into LDS partials; waves 0,1 reduce and store.
  v8f acc = {};
  if (wave < 4) {
    int ntile = wave & 1;
    const f16* Asrc = (wave < 2) ? &A16[0][0] : &AP16[0][0];
    const f16* Bsrc = (wave < 2) ? (Vh + (size_t)h * DKDIM * NPOS) : RVh;
    for (int kc = 0; kc < 32; ++kc) {
      v16h aa = load_a(Asrc, 0, NPOS, kc * 32, lane);
      v16h bb = load_b(Bsrc, ntile * 16, NPOS, kc * 32, lane);
      acc = wmma_f16(aa, bb, acc);
    }
    if (wave >= 2) {
#pragma unroll
      for (int g = 0; g < 8; ++g) Opart[ntile][g + 8 * s][n] = acc[g];
    }
  }
  __syncthreads();
  if (wave < 2) {
    int d = wave * 16 + n;
#pragma unroll
    for (int g = 0; g < 8; ++g) {
      int i = i0 + g + 8 * s;
      float o = acc[g] + Opart[wave][g + 8 * s][n];
      att_t[(size_t)i * 256 + h * DKDIM + d] = (f16)o;
    }
  }
}

// out = W0(128x256) @ att(256x1024) + h  (residual), f32 output
__global__ __launch_bounds__(256) void oproj_kernel(const f16* W0h,
                                                    const f16* att_t,
                                                    const float* hres,
                                                    float* out) {
  int tileId = blockIdx.x * 8 + (threadIdx.x >> 5);  // 512 tiles (8 x 64)
  int lane = threadIdx.x & 31;
  int m0 = (tileId >> 6) * 16, i0 = (tileId & 63) * 16;
  v8f acc = {};
#pragma unroll
  for (int kc = 0; kc < 8; ++kc) {
    v16h a = load_a(W0h, m0, 256, kc * 32, lane);
    v16h b = load_b(att_t, i0, 256, kc * 32, lane);
    acc = wmma_f16(a, b, acc);
  }
  int n = lane & 15, s = lane >> 4;
  int i = i0 + n;
#pragma unroll
  for (int g = 0; g < 8; ++g) {
    int co = m0 + g + 8 * s;
    out[co * NPOS + i] = acc[g] + hres[co * NPOS + i];
  }
}

// ---------------- host launch ----------------
extern "C" void kernel_launch(void* const* d_in, const int* in_sizes, int n_in,
                              void* d_out, int out_size, void* d_ws,
                              size_t ws_size, hipStream_t stream) {
  const float* x = (const float*)d_in[0];
  const float* We = (const float*)d_in[1];
  const float* be = (const float*)d_in[2];
  const float* Wq = (const float*)d_in[3];
  const float* Wkd = (const float*)d_in[4];
  const float* Wkp = (const float*)d_in[5];
  const float* Wvd = (const float*)d_in[6];
  const float* Wvp = (const float*)d_in[7];
  const float* RQ = (const float*)d_in[8];
  const float* RK = (const float*)d_in[9];
  const float* RV = (const float*)d_in[10];
  const float* W0 = (const float*)d_in[11];
  float* out = (float*)d_out;

  char* ws = (char*)d_ws;
  size_t off = 0;
  auto alloc = [&](size_t bytes) -> void* {
    void* p = ws + off;
    off += (bytes + 255) & ~(size_t)255;
    return p;
  };

  float* gx = (float*)alloc(CDIM * NPOS * 4);
  float* h = (float*)alloc(CDIM * NPOS * 4);
  float* hk = (float*)alloc(CDIM * NPOS * 4);
  float* hv = (float*)alloc(CDIM * NPOS * 4);
  f16* h2t = (f16*)alloc(NPOS * CDIM * 2);
  f16* Qt = (f16*)alloc((size_t)HEADS * NPOS * DKDIM * 2);
  f16* Kt = (f16*)alloc((size_t)HEADS * NPOS * DKDIM * 2);
  f16* Vh = (f16*)alloc((size_t)HEADS * DKDIM * NPOS * 2);
  f16* RQt = (f16*)alloc(DKDIM * NPOS * 2);
  f16* RKt = (f16*)alloc(DKDIM * NPOS * 2);
  f16* RVh = (f16*)alloc(DKDIM * NPOS * 2);
  f16* Wqh = (f16*)alloc(256 * CDIM * 2);
  f16* W0h = (f16*)alloc(CDIM * 256 * 2);
  f16* att_t = (f16*)alloc((size_t)NPOS * 256 * 2);
  f16* TK = (f16*)alloc((size_t)HEADS * NPOS * NPOS * 2);  // 16 MB
  f16* TQ = (f16*)alloc((size_t)HEADS * NPOS * NPOS * 2);  // 16 MB

  cvt_kernel<<<128, 256, 0, stream>>>(RQ, RK, RV, Wq, W0, RQt, RKt, RVh, Wqh,
                                      W0h);
  gelu_kernel<<<512, 256, 0, stream>>>(x, gx);
  embed_kernel<<<512, 256, 0, stream>>>(gx, We, be, x, h, h2t);
  dw_kernel<<<512, 256, 0, stream>>>(h, Wkd, Wvd, hk, hv);
  pw_kernel<<<1024, 256, 0, stream>>>(hk, hv, Wkp, Wvp, Kt, Vh);
  qproj_kernel<<<128, 256, 0, stream>>>(Wqh, h2t, Qt);
  tgemm_kernel<<<4096, 256, 0, stream>>>(Qt, RKt, TK);  // TK = Q^T @ RK
  tgemm_kernel<<<4096, 256, 0, stream>>>(Kt, RQt, TQ);  // TQ = K^T @ RQ
  attn_kernel<<<512, 256, 0, stream>>>(Qt, Kt, Vh, TK, TQ, RVh, att_t);
  oproj_kernel<<<64, 256, 0, stream>>>(W0h, att_t, h, out);
}
